// GcnGatModel1_11793980194921
// MI455X (gfx1250) — matrix operand order repaired and checked by reading, not compiled, not run on hardware
//
#include <hip/hip_runtime.h>
#include <hip/hip_bf16.h>
#include <cstdint>

// ---------------- problem constants (match reference) ----------------
#define B_    64
#define N_    1024
#define E_    16384
#define E2_   (E_ + N_)      // edges + self loops
#define FIN_  300
#define FOUT_ 8
#define BETA_ 0.04879016416943205f   // log(0.1/2 + 1.0)

typedef __attribute__((ext_vector_type(16))) _Float16 v16h;
typedef __attribute__((ext_vector_type(8)))  float    v8f;

static __device__ __forceinline__ float lrelu02(float x) { return x > 0.f ? x : 0.2f * x; }

static __device__ void atomicMaxF(float* addr, float v) {
  int* ai = (int*)addr;
  int old = __float_as_int(*addr);
  while (__int_as_float(old) < v) {
    int assumed = old;
    old = atomicCAS(ai, assumed, __float_as_int(v));
    if (old == assumed) break;
  }
}

// ---------------- GCN normalization prep ----------------
__global__ void k_deg_init(float* __restrict__ degw, float* __restrict__ degu) {
  int idx = blockIdx.x * blockDim.x + threadIdx.x;
  if (idx >= B_ * N_) return;
  degw[idx] = 1.0f;   // self-loop weight
  degu[idx] = 1.0f;
}

__global__ void k_deg_edges(const int* __restrict__ ei, const float* __restrict__ ew,
                            float* __restrict__ degw, float* __restrict__ degu) {
  int idx = blockIdx.x * blockDim.x + threadIdx.x;
  if (idx >= B_ * E_) return;
  int g = idx / E_, e = idx % E_;
  int d = ei[(size_t)g * 2 * E_ + E_ + e];
  atomicAdd(&degw[g * N_ + d], ew[(size_t)g * E_ + e]);
  atomicAdd(&degu[g * N_ + d], 1.0f);
}

__global__ void k_dinv(float* __restrict__ degw, float* __restrict__ degu) {
  int idx = blockIdx.x * blockDim.x + threadIdx.x;
  if (idx >= B_ * N_) return;
  degw[idx] = rsqrtf(fmaxf(degw[idx], 1e-12f));
  degu[idx] = rsqrtf(fmaxf(degu[idx], 1e-12f));
}

__global__ void k_wnorm(const int* __restrict__ ei, const float* __restrict__ ew,
                        const float* __restrict__ dw, const float* __restrict__ du,
                        float* __restrict__ wN, float* __restrict__ w1N) {
  int idx = blockIdx.x * blockDim.x + threadIdx.x;
  if (idx >= B_ * E2_) return;
  int g = idx / E2_, e = idx % E2_;
  if (e < E_) {
    int s = ei[(size_t)g * 2 * E_ + e];
    int d = ei[(size_t)g * 2 * E_ + E_ + e];
    wN[idx]  = dw[g * N_ + s] * ew[(size_t)g * E_ + e] * dw[g * N_ + d];
    w1N[idx] = du[g * N_ + s] * du[g * N_ + d];
  } else {
    int i = e - E_;
    float a = dw[g * N_ + i]; wN[idx]  = a * a;
    float b = du[g * N_ + i]; w1N[idx] = b * b;
  }
}

// ---------------- WMMA GEMM: Out[B,N,dout] = Hin[B,N,din] @ W[din,dout] ----------------
// Weight pre-swizzle into the exact per-lane V_WMMA_F32_16X16X32_F16 B-operand layout:
// tile (kt,ct) -> 512 halves; lane L (hi=L>>4, lo=L&15) holds col (ct*16+lo),
// rows kt*32 + (j<8 ? hi*8+j : 16+hi*8+(j-8)) for j=0..15, stored contiguously -> one 32B load per tile.
__global__ void k_prep_w(const float* __restrict__ W, _Float16* __restrict__ Wsw,
                         int din, int dout, int ktiles) {
  int idx = blockIdx.x * blockDim.x + threadIdx.x;
  int ctiles = dout >> 4;
  int total = ktiles * ctiles * 512;
  if (idx >= total) return;
  int tile = idx >> 9, r = idx & 511;
  int lane = r >> 4, j = r & 15;
  int hi = lane >> 4, lo = lane & 15;
  int kt = tile / ctiles, ct = tile % ctiles;
  int k = kt * 32 + ((j < 8) ? (hi * 8 + j) : (16 + hi * 8 + (j - 8)));
  int c = ct * 16 + lo;
  float v = (k < din) ? W[(size_t)k * dout + c] : 0.f;
  Wsw[idx] = (_Float16)v;
}

// A-operand for one full 32-K tile: two runs of 8 consecutive floats per lane
// (k0+hi*8 .. +7 and k0+16+hi*8 .. +7) -> 4 unguarded b128 loads + cvt.
static __device__ __forceinline__ v16h loadA32(const float* __restrict__ arow, int k0, int hi) {
  const float4* p0 = (const float4*)(arow + k0 + hi * 8);
  const float4* p1 = (const float4*)(arow + k0 + 16 + hi * 8);
  float4 a0 = p0[0], a1 = p0[1];
  float4 a2 = p1[0], a3 = p1[1];
  v16h a;
  a[0] = (_Float16)a0.x; a[1] = (_Float16)a0.y; a[2]  = (_Float16)a0.z; a[3]  = (_Float16)a0.w;
  a[4] = (_Float16)a1.x; a[5] = (_Float16)a1.y; a[6]  = (_Float16)a1.z; a[7]  = (_Float16)a1.w;
  a[8] = (_Float16)a2.x; a[9] = (_Float16)a2.y; a[10] = (_Float16)a2.z; a[11] = (_Float16)a2.w;
  a[12] = (_Float16)a3.x; a[13] = (_Float16)a3.y; a[14] = (_Float16)a3.z; a[15] = (_Float16)a3.w;
  return a;
}

// One wave -> 16 rows x 2 column-tiles (16x32 of C). Software-pipelined K loop.
__global__ void k_wmma_gemm(const float* __restrict__ Hin, const _Float16* __restrict__ Wsw,
                            float* __restrict__ Out, int din, int dout, int kfull, int ktiles) {
  const int lane = threadIdx.x;           // wave32
  const int hi = lane >> 4, lo = lane & 15;
  const int ct0 = blockIdx.y * 2;
  const int ctiles = dout >> 4;
  const size_t gbase = (size_t)blockIdx.z * N_;
  const int row0 = blockIdx.x * 16;
  const float* __restrict__ arow = Hin + (gbase + row0 + lo) * (size_t)din;
  const _Float16* __restrict__ wp = Wsw + (size_t)ct0 * 512 + (lane << 4);
  const size_t wstride = (size_t)ctiles * 512;

  v8f acc0 = {}, acc1 = {};
  v16h a  = loadA32(arow, 0, hi);
  v16h b0 = *(const v16h*)(wp);
  v16h b1 = *(const v16h*)(wp + 512);
  for (int kt = 1; kt < kfull; ++kt) {
    v16h an  = loadA32(arow, kt * 32, hi);
    v16h bn0 = *(const v16h*)(wp + (size_t)kt * wstride);
    v16h bn1 = *(const v16h*)(wp + (size_t)kt * wstride + 512);
    acc0 = __builtin_amdgcn_wmma_f32_16x16x32_f16(false, a, false, b0, (short)0, acc0, false, false);
    acc1 = __builtin_amdgcn_wmma_f32_16x16x32_f16(false, a, false, b1, (short)0, acc1, false, false);
    a = an; b0 = bn0; b1 = bn1;
  }
  acc0 = __builtin_amdgcn_wmma_f32_16x16x32_f16(false, a, false, b0, (short)0, acc0, false, false);
  acc1 = __builtin_amdgcn_wmma_f32_16x16x32_f16(false, a, false, b1, (short)0, acc1, false, false);

  if (kfull < ktiles) {           // ragged K tail (only din=300): clamped loads + selects
    const int k0 = kfull * 32;
    v16h at;
#pragma unroll
    for (int j = 0; j < 16; ++j) {
      int kk = k0 + ((j < 8) ? (hi * 8 + j) : (16 + hi * 8 + (j - 8)));
      float v = arow[kk < din ? kk : 0];
      at[j] = (_Float16)(kk < din ? v : 0.f);
    }
    v16h bt0 = *(const v16h*)(wp + (size_t)kfull * wstride);        // zero-padded by prep
    v16h bt1 = *(const v16h*)(wp + (size_t)kfull * wstride + 512);
    acc0 = __builtin_amdgcn_wmma_f32_16x16x32_f16(false, at, false, bt0, (short)0, acc0, false, false);
    acc1 = __builtin_amdgcn_wmma_f32_16x16x32_f16(false, at, false, bt1, (short)0, acc1, false, false);
  }

  // C layout: VGPR i -> row (row0 + hi*8 + i), col (ct*16 + lo)
#pragma unroll
  for (int i = 0; i < 8; ++i) {
    size_t rbase = (gbase + row0 + hi * 8 + i) * (size_t)dout;
    Out[rbase + ct0 * 16 + lo]      = acc0[i];
    Out[rbase + ct0 * 16 + 16 + lo] = acc1[i];
  }
}

// ---------------- edge propagation: out[c] += w * h[r], self loops fused ----------------
__global__ void k_prop_self4(const float* __restrict__ h, const float* __restrict__ wN,
                             float* __restrict__ out, int D) {
  int D4 = D >> 2;
  int idx = blockIdx.x * blockDim.x + threadIdx.x;
  if (idx >= B_ * N_ * D4) return;
  int g = idx / (N_ * D4), r = idx % (N_ * D4);
  int i = r / D4, f = (r % D4) * 4;
  float w = wN[(size_t)g * E2_ + E_ + i];
  const float* hp = h + ((size_t)(g * N_ + i)) * D + f;
  float* op = out + ((size_t)(g * N_ + i)) * D + f;
  op[0] = w * hp[0]; op[1] = w * hp[1]; op[2] = w * hp[2]; op[3] = w * hp[3];
}

__global__ void k_prop_edge4(const float* __restrict__ h, const int* __restrict__ ei,
                             const float* __restrict__ wN, float* __restrict__ out, int D) {
  int D4 = D >> 2;
  int idx = blockIdx.x * blockDim.x + threadIdx.x;
  if (idx >= B_ * E_ * D4) return;
  int g = idx / (E_ * D4), r = idx % (E_ * D4);
  int e = r / D4, f = (r % D4) * 4;
  int s = ei[(size_t)g * 2 * E_ + e];
  int d = ei[(size_t)g * 2 * E_ + E_ + e];
  float w = wN[(size_t)g * E2_ + e];
  const float* hp = h + ((size_t)(g * N_ + s)) * D + f;
  float* op = out + ((size_t)(g * N_ + d)) * D + f;
  atomicAdd(op + 0, w * hp[0]); atomicAdd(op + 1, w * hp[1]);
  atomicAdd(op + 2, w * hp[2]); atomicAdd(op + 3, w * hp[3]);
}

// scalar variants for D==5 (pool assignment)
__global__ void k_prop_self1(const float* __restrict__ h, const float* __restrict__ wN,
                             float* __restrict__ out, int D) {
  int idx = blockIdx.x * blockDim.x + threadIdx.x;
  if (idx >= B_ * N_ * D) return;
  int g = idx / (N_ * D), r = idx % (N_ * D);
  int i = r / D, f = r % D;
  out[((size_t)(g * N_ + i)) * D + f] = wN[(size_t)g * E2_ + E_ + i] * h[((size_t)(g * N_ + i)) * D + f];
}

__global__ void k_prop_edge1(const float* __restrict__ h, const int* __restrict__ ei,
                             const float* __restrict__ wN, float* __restrict__ out, int D) {
  int idx = blockIdx.x * blockDim.x + threadIdx.x;
  if (idx >= B_ * E_ * D) return;
  int g = idx / (E_ * D), r = idx % (E_ * D);
  int e = r / D, f = r % D;
  int s = ei[(size_t)g * 2 * E_ + e];
  int d = ei[(size_t)g * 2 * E_ + E_ + e];
  atomicAdd(&out[((size_t)(g * N_ + d)) * D + f],
            wN[(size_t)g * E2_ + e] * h[((size_t)(g * N_ + s)) * D + f]);
}

// ---------------- elementwise ----------------
__global__ void k_bias_act(float* __restrict__ x, const float* __restrict__ b, int D, int do_relu) {
  int idx = blockIdx.x * blockDim.x + threadIdx.x;
  if (idx >= B_ * N_ * D) return;
  float v = x[idx] + b[idx % D];
  x[idx] = do_relu ? fmaxf(v, 0.f) : v;
}

__global__ void k_zcomb(const float* __restrict__ p, const float* __restrict__ x0,
                        float* __restrict__ z, int D) {  // z = (1-alpha)*prop + alpha*x0, alpha=0.5
  int idx = blockIdx.x * blockDim.x + threadIdx.x;
  if (idx >= B_ * N_ * D) return;
  z[idx] = 0.5f * (p[idx] + x0[idx]);
}

__global__ void k_gcn2fin(const float* __restrict__ z, const float* __restrict__ zw,
                          float* __restrict__ out, int D) {  // relu((1-beta)z + beta*(z@W))
  int idx = blockIdx.x * blockDim.x + threadIdx.x;
  if (idx >= B_ * N_ * D) return;
  out[idx] = fmaxf((1.0f - BETA_) * z[idx] + BETA_ * zw[idx], 0.f);
}

// ---------------- GATv2 (heads=4, hdim=32, concat, self loops) ----------------
__global__ void k_mx_init(float* __restrict__ mx) {
  int idx = blockIdx.x * blockDim.x + threadIdx.x;
  if (idx < B_ * N_ * 4) mx[idx] = -3.0e38f;
}

__global__ void k_att_logits(const float* __restrict__ xl, const float* __restrict__ xr,
                             const int* __restrict__ ei, const float* __restrict__ att,
                             float* __restrict__ lg, float* __restrict__ mx) {
  int idx = blockIdx.x * blockDim.x + threadIdx.x;
  if (idx >= B_ * E2_ * 4) return;
  int g = idx / (E2_ * 4), r = idx % (E2_ * 4);
  int e = r >> 2, hd = r & 3;
  int s, d;
  if (e < E_) { s = ei[(size_t)g * 2 * E_ + e]; d = ei[(size_t)g * 2 * E_ + E_ + e]; }
  else        { s = d = e - E_; }
  const float* pl = xl + ((size_t)(g * N_ + s)) * 128 + hd * 32;
  const float* pr = xr + ((size_t)(g * N_ + d)) * 128 + hd * 32;
  const float* pa = att + hd * 32;
  float acc = 0.f;
#pragma unroll
  for (int c = 0; c < 32; ++c) acc += pa[c] * lrelu02(pl[c] + pr[c]);
  lg[idx] = acc;
  atomicMaxF(&mx[((size_t)(g * N_ + d)) * 4 + hd], acc);
}

__global__ void k_att_exp(const int* __restrict__ ei, float* __restrict__ lg,
                          const float* __restrict__ mx, float* __restrict__ ss) {
  int idx = blockIdx.x * blockDim.x + threadIdx.x;
  if (idx >= B_ * E2_ * 4) return;
  int g = idx / (E2_ * 4), r = idx % (E2_ * 4);
  int e = r >> 2, hd = r & 3;
  int d = (e < E_) ? ei[(size_t)g * 2 * E_ + E_ + e] : (e - E_);
  float a = __expf(lg[idx] - mx[((size_t)(g * N_ + d)) * 4 + hd]);
  lg[idx] = a;
  atomicAdd(&ss[((size_t)(g * N_ + d)) * 4 + hd], a);
}

__global__ void k_att_scatter(const float* __restrict__ xl, const int* __restrict__ ei,
                              const float* __restrict__ lg, const float* __restrict__ ss,
                              float* __restrict__ out) {
  int idx = blockIdx.x * blockDim.x + threadIdx.x;
  if (idx >= B_ * E2_ * 4) return;
  int g = idx / (E2_ * 4), r = idx % (E2_ * 4);
  int e = r >> 2, hd = r & 3;
  int s, d;
  if (e < E_) { s = ei[(size_t)g * 2 * E_ + e]; d = ei[(size_t)g * 2 * E_ + E_ + e]; }
  else        { s = d = e - E_; }
  float alpha = lg[idx] / ss[((size_t)(g * N_ + d)) * 4 + hd];
  const float* pl = xl + ((size_t)(g * N_ + s)) * 128 + hd * 32;
  float* po = out + ((size_t)(g * N_ + d)) * 128 + hd * 32;
#pragma unroll
  for (int c = 0; c < 32; ++c) atomicAdd(po + c, alpha * pl[c]);
}

// ---------------- pooling ----------------
__global__ void k_sgemm5(const float* __restrict__ H, const float* __restrict__ W,
                         float* __restrict__ out) {  // [B,N,256] @ [256,5]
  int idx = blockIdx.x * blockDim.x + threadIdx.x;
  if (idx >= B_ * N_ * 5) return;
  int g = idx / (N_ * 5), r = idx % (N_ * 5);
  int i = r / 5, o = r % 5;
  const float* hp = H + ((size_t)(g * N_ + i)) * 256;
  float acc = 0.f;
  for (int k = 0; k < 256; ++k) acc += hp[k] * W[k * 5 + o];
  out[idx] = acc;
}

__global__ void k_softmax5(float* __restrict__ s, const float* __restrict__ bias) {
  int idx = blockIdx.x * blockDim.x + threadIdx.x;
  if (idx >= B_ * N_) return;
  float* p = s + (size_t)idx * 5;
  float v[5], m = -3.0e38f;
#pragma unroll
  for (int j = 0; j < 5; ++j) { v[j] = p[j] + bias[j]; m = fmaxf(m, v[j]); }
  float sum = 0.f;
#pragma unroll
  for (int j = 0; j < 5; ++j) { v[j] = __expf(v[j] - m); sum += v[j]; }
  float inv = 1.0f / sum;
#pragma unroll
  for (int j = 0; j < 5; ++j) p[j] = v[j] * inv;
}

__global__ void k_nodes(const float* __restrict__ s, const float* __restrict__ h,
                        float* __restrict__ nd) {  // nodes = s^T @ h, [5,256] per graph
  int g = blockIdx.x, f = threadIdx.x;  // 256 threads
  float acc[5] = {0.f, 0.f, 0.f, 0.f, 0.f};
  for (int i = 0; i < N_; ++i) {
    float hv = h[((size_t)(g * N_ + i)) * 256 + f];
    const float* sp = s + (size_t)(g * N_ + i) * 5;
#pragma unroll
    for (int p = 0; p < 5; ++p) acc[p] += sp[p] * hv;
  }
#pragma unroll
  for (int p = 0; p < 5; ++p) nd[((size_t)g * 5 + p) * 256 + f] = acc[p];
}

// DenseGCNConv output feeds softmax(..., axis=-1) on a [5,1] tensor, which is exactly
// 1.0 for every element. So diff-pool stage 2 reduces to pooled[f] = sum_p nodes[p,f];
// adj2 / Wp2 / bp2 provably cannot affect the result.
__global__ void k_pool(const float* __restrict__ nd, float* __restrict__ pl) {
  int g = blockIdx.x, f = threadIdx.x;
  float a = 0.f;
#pragma unroll
  for (int p = 0; p < 5; ++p) a += nd[((size_t)g * 5 + p) * 256 + f];
  pl[g * 256 + f] = a;
}

__global__ void k_out(const float* __restrict__ pl, const float* __restrict__ Wo,
                      const float* __restrict__ bo, float* __restrict__ out) {
  int idx = blockIdx.x * blockDim.x + threadIdx.x;
  if (idx >= B_ * FOUT_) return;
  int g = idx / FOUT_, o = idx % FOUT_;
  float a = bo[o];
  for (int f = 0; f < 256; ++f) a += pl[g * 256 + f] * Wo[f * FOUT_ + o];
  out[idx] = a;
}

// ---------------- host orchestration ----------------
static inline int cdiv(long long a, int b) { return (int)((a + b - 1) / b); }

extern "C" void kernel_launch(void* const* d_in, const int* in_sizes, int n_in,
                              void* d_out, int out_size, void* d_ws, size_t ws_size,
                              hipStream_t stream) {
  (void)in_sizes; (void)n_in; (void)out_size; (void)ws_size;
  const float* x   = (const float*)d_in[0];
  const int*   ei  = (const int*)  d_in[1];
  const float* ew  = (const float*)d_in[2];
  const float* W1  = (const float*)d_in[3];  const float* b1 = (const float*)d_in[4];
  const float* W2  = (const float*)d_in[5];  const float* b2 = (const float*)d_in[6];
  const float* W3  = (const float*)d_in[7];  const float* b3 = (const float*)d_in[8];
  const float* W4  = (const float*)d_in[9];  const float* b4 = (const float*)d_in[10];
  const float* Wl  = (const float*)d_in[11]; const float* Wr = (const float*)d_in[12];
  const float* att = (const float*)d_in[13]; const float* bg = (const float*)d_in[14];
  const float* Wg1 = (const float*)d_in[15];
  const float* Wm  = (const float*)d_in[16]; const float* bm = (const float*)d_in[17];
  const float* Wg2 = (const float*)d_in[18];
  const float* Wp1 = (const float*)d_in[19]; const float* bp1 = (const float*)d_in[20];
  // d_in[21]=Wp2, d_in[22]=bp2 unused: nullified by softmax over a size-1 axis (see k_pool)
  const float* Wo  = (const float*)d_in[23]; const float* bo = (const float*)d_in[24];
  float* out = (float*)d_out;

  float* ws = (float*)d_ws;
  size_t off = 0;
  auto take = [&](size_t nfl) { float* p = ws + off; off += (nfl + 63) & ~(size_t)63; return p; };
  float* degw = take((size_t)B_ * N_);
  float* degu = take((size_t)B_ * N_);
  float* wN   = take((size_t)B_ * E2_);
  float* w1N  = take((size_t)B_ * E2_);
  float* A    = take((size_t)B_ * N_ * 256);
  float* Bb   = take((size_t)B_ * N_ * 256);
  float* H1   = take((size_t)B_ * N_ * 256);
  float* H2   = take((size_t)B_ * N_ * 128);
  float* Z    = take((size_t)B_ * N_ * 256);
  float* LG   = take((size_t)B_ * E2_ * 4);
  float* MX   = take((size_t)B_ * N_ * 4);
  float* SS   = take((size_t)B_ * N_ * 4);
  float* SA   = take((size_t)B_ * N_ * 5);
  float* SB   = take((size_t)B_ * N_ * 5);
  float* ND   = take((size_t)B_ * 5 * 256);
  float* PL   = take((size_t)B_ * 256);
  _Float16* WSW = (_Float16*)take(49152);  // 98304 halves; max layer needs 81920
  float* XL = A;                               // reuse A as two [B,N,128] halves
  float* XR = A + (size_t)B_ * N_ * 128;

  const int T = 256;
  auto gemm = [&](const float* Hin, const float* W, float* Out, int din, int dout) {
    int ktiles = (din + 31) / 32, kfull = din / 32, ctiles = dout / 16;
    int prep = ktiles * ctiles * 512;
    k_prep_w<<<cdiv(prep, T), T, 0, stream>>>(W, WSW, din, dout, ktiles);
    k_wmma_gemm<<<dim3(N_ / 16, ctiles / 2, B_), dim3(32), 0, stream>>>(Hin, WSW, Out, din, dout, kfull, ktiles);
  };
  auto prop4 = [&](const float* h, const float* w, float* o, int D) {
    k_prop_self4<<<cdiv((long long)B_ * N_ * (D / 4), T), T, 0, stream>>>(h, w, o, D);
    k_prop_edge4<<<cdiv((long long)B_ * E_ * (D / 4), T), T, 0, stream>>>(h, ei, w, o, D);
  };
  auto bias_act = [&](float* xp, const float* bp, int D, int relu) {
    k_bias_act<<<cdiv((long long)B_ * N_ * D, T), T, 0, stream>>>(xp, bp, D, relu);
  };

  // --- GCN normalization (weighted + unweighted) ---
  k_deg_init<<<cdiv(B_ * N_, T), T, 0, stream>>>(degw, degu);
  k_deg_edges<<<cdiv(B_ * E_, T), T, 0, stream>>>(ei, ew, degw, degu);
  k_dinv<<<cdiv(B_ * N_, T), T, 0, stream>>>(degw, degu);
  k_wnorm<<<cdiv((long long)B_ * E2_, T), T, 0, stream>>>(ei, ew, degw, degu, wN, w1N);

  // --- GCN stack ---
  gemm(x, W1, A, FIN_, 256);  prop4(A, wN, H1, 256);  bias_act(H1, b1, 256, 1);
  gemm(H1, W2, A, 256, 128);  prop4(A, wN, H2, 128);  bias_act(H2, b2, 128, 1);
  gemm(H2, W3, A, 128, 64);   prop4(A, wN, Bb, 64);   bias_act(Bb, b3, 64, 1);
  gemm(Bb, W4, A, 64, 32);    prop4(A, wN, Bb, 32);   bias_act(Bb, b4, 32, 1);   // Bb = h [N,32]

  // --- GATv2 ---
  gemm(Bb, Wl, XL, 32, 128);
  gemm(Bb, Wr, XR, 32, 128);
  k_mx_init<<<cdiv(B_ * N_ * 4, T), T, 0, stream>>>(MX);
  hipMemsetAsync(SS, 0, (size_t)B_ * N_ * 4 * sizeof(float), stream);
  k_att_logits<<<cdiv((long long)B_ * E2_ * 4, T), T, 0, stream>>>(XL, XR, ei, att, LG, MX);
  k_att_exp<<<cdiv((long long)B_ * E2_ * 4, T), T, 0, stream>>>(ei, LG, MX, SS);
  hipMemsetAsync(Z, 0, (size_t)B_ * N_ * 128 * sizeof(float), stream);
  k_att_scatter<<<cdiv((long long)B_ * E2_ * 4, T), T, 0, stream>>>(XL, ei, LG, SS, Z);
  bias_act(Z, bg, 128, 1);                                                       // Z = h [N,128]

  // --- GCN2Conv #1 (x0 = h2) ---
  prop4(Z, wN, XL, 128);
  k_zcomb<<<cdiv((long long)B_ * N_ * 128, T), T, 0, stream>>>(XL, H2, XR, 128);
  gemm(XR, Wg1, Bb, 128, 128);
  k_gcn2fin<<<cdiv((long long)B_ * N_ * 128, T), T, 0, stream>>>(XR, Bb, Z, 128); // Z = h [N,128]

  // --- GCNConv(128,256) with unweighted norm ---
  gemm(Z, Wm, A, 128, 256);
  prop4(A, w1N, Bb, 256);  bias_act(Bb, bm, 256, 1);                              // Bb = h [N,256]

  // --- GCN2Conv #2 (x0 = h1) ---
  prop4(Bb, wN, A, 256);
  k_zcomb<<<cdiv((long long)B_ * N_ * 256, T), T, 0, stream>>>(A, H1, Z, 256);
  gemm(Z, Wg2, A, 256, 256);
  k_gcn2fin<<<cdiv((long long)B_ * N_ * 256, T), T, 0, stream>>>(Z, A, Bb, 256);  // Bb = h [N,256]

  // --- diff-pool ---
  k_sgemm5<<<cdiv((long long)B_ * N_ * 5, T), T, 0, stream>>>(Bb, Wp1, SA);
  k_prop_self1<<<cdiv((long long)B_ * N_ * 5, T), T, 0, stream>>>(SA, wN, SB, 5);
  k_prop_edge1<<<cdiv((long long)B_ * E_ * 5, T), T, 0, stream>>>(SA, ei, wN, SB, 5);
  k_softmax5<<<cdiv(B_ * N_, T), T, 0, stream>>>(SB, bp1);
  k_nodes<<<dim3(B_), dim3(256), 0, stream>>>(SB, Bb, ND);
  k_pool<<<dim3(B_), dim3(256), 0, stream>>>(ND, PL);
  k_out<<<cdiv(B_ * FOUT_, T), T, 0, stream>>>(PL, Wo, bo, out);
}
